// NetGINE_30502857736593
// MI455X (gfx1250) — matrix operand order repaired and checked by reading, not compiled, not run on hardware
//
#include <hip/hip_runtime.h>
#include <hip/hip_bf16.h>

#define DIMC   64
#define NFC    28
#define EFC    3
#define NNODES 100000
#define NEDGES 1200000
#define NGRAPH 1000

typedef __attribute__((ext_vector_type(16))) _Float16 v16h;
typedef __attribute__((ext_vector_type(8)))  float    v8f;
typedef __attribute__((ext_vector_type(4)))  unsigned u32x4;
typedef __attribute__((ext_vector_type(4)))  float    f32x4;

// ---------------------------------------------------------------------------
// Packed WMMA fragment loaders (wave32, v_wmma_f32_16x16x32_f16).
// A tile lives in LDS row-major as f16 with LDW=36 dwords (72 halves) per row.
// A-fragment dwords 0..3 = K pairs {8h+0..7}, dwords 4..7 = {16+8h+0..7}:
// two contiguous 16B runs -> two ds_load_b128 per fragment.
// B is staged pre-packed: swb[pair*LDB + n] = {B[2p][n], B[2p+1][n]} so each
// fragment dword is one ds_load_b32 (no repacking); LDB=68 de-conflicts the
// two lane halves (4*68 % 64 == 16).
// ---------------------------------------------------------------------------
#define LDW 36
#define LDB 68

__device__ __forceinline__ v16h frag_a_pk(const unsigned* base32, int lane) {
  const int m = lane & 15, half = lane >> 4;
  union { u32x4 q[2]; v16h h; } c;
  c.q[0] = *(const u32x4*)(base32 + m * LDW + 4 * half);
  c.q[1] = *(const u32x4*)(base32 + m * LDW + 8 + 4 * half);
  return c.h;
}

__device__ __forceinline__ v16h frag_b_pk(const unsigned* swb, int ncol, int lane) {
  const int col = (lane & 15) + ncol, half = lane >> 4;
  union { unsigned u[8]; v16h h; } c;
#pragma unroll
  for (int v = 0; v < 8; ++v) c.u[v] = swb[(v + 4 * half + (v & 4)) * LDB + col];
  return c.h;
}

// stage 64x64 f32 weight (row-major [K][N]) into packed K-pair LDS layout
__device__ __forceinline__ void stage_w_packed(const float* __restrict__ w,
                                               unsigned* swb, int t) {
  for (int i = t; i < 32 * 64; i += 256) {
    const int p = i >> 6, n = i & 63;
    union { _Float16 h[2]; unsigned u; } pk;
    pk.h[0] = (_Float16)w[(2 * p) * 64 + n];
    pk.h[1] = (_Float16)w[(2 * p + 1) * 64 + n];
    swb[p * LDB + n] = pk.u;
  }
}

// ---------------------------------------------------------------------------
// Layers 1,2: fused edge MLP (WMMA) + message + scatter-add.
// 128 edges/block, 256 threads (8 waves). Never materializes E x 64 tensor.
// ---------------------------------------------------------------------------
#define TILE_E 128

__global__ __launch_bounds__(256)
void edge_fused_wmma(const float* __restrict__ hin,        // N x 64 node feats
                     const float* __restrict__ eattr,      // E x 3
                     const float* __restrict__ ew,         // E
                     const long long* __restrict__ rowi,
                     const long long* __restrict__ coli,
                     const float* __restrict__ w1,         // 3 x 64
                     const float* __restrict__ b1,         // 64
                     const float* __restrict__ w2,         // 64 x 64
                     const float* __restrict__ b2,         // 64
                     float* __restrict__ agg)              // N x 64 (atomic)
{
  __shared__ unsigned sh32[TILE_E * LDW];     // hidden tile, packed f16 pairs
  __shared__ unsigned swb[32 * LDB];          // W2, packed K-pair layout
  __shared__ float    sea[TILE_E * 3];
  __shared__ unsigned srow[TILE_E], scol[TILE_E];   // unsigned: no sign-extend
  __shared__ float    sew[TILE_E];
  __shared__ float    sw1[3 * 64];
  __shared__ float    sb1[64], sb2[64];

  const int t = threadIdx.x;
  const int e0 = blockIdx.x * TILE_E;

  for (int i = t; i < TILE_E; i += 256) {
    const int e = e0 + i;
    srow[i] = (unsigned)rowi[e] * 64u;    // pre-scaled row offsets
    scol[i] = (unsigned)coli[e] * 64u;
    sew[i]  = ew[e];
  }
  for (int i = t; i < TILE_E * 3; i += 256) sea[i] = eattr[e0 * 3 + i];
  for (int i = t; i < 3 * 64;    i += 256) sw1[i] = w1[i];
  if (t < 64) { sb1[t] = b1[t]; sb2[t] = b2[t]; }
  stage_w_packed(w2, swb, t);
  __syncthreads();

  // hidden = relu(ea @ W1 + b1) (128 x 64), packed two channels per dword
  for (int i = t; i < TILE_E * 32; i += 256) {
    const int e = i >> 5, cp = i & 31;
    const float a0 = sea[e * 3 + 0], a1 = sea[e * 3 + 1], a2 = sea[e * 3 + 2];
    union { _Float16 h[2]; unsigned u; } pk;
#pragma unroll
    for (int s = 0; s < 2; ++s) {
      const int c = 2 * cp + s;
      float h = sb1[c] + a0 * sw1[c] + a1 * sw1[64 + c] + a2 * sw1[128 + c];
      pk.h[s] = (_Float16)fmaxf(h, 0.f);
    }
    sh32[e * LDW + cp] = pk.u;
  }
  __syncthreads();

  const int wave = t >> 5, lane = t & 31;
  const int half = lane >> 4, nlane = lane & 15;

#pragma unroll
  for (int nt = 0; nt < 4; ++nt) {
    const float bias = sb2[nt * 16 + nlane];
    v8f acc = {bias, bias, bias, bias, bias, bias, bias, bias};
#pragma unroll
    for (int kt = 0; kt < 2; ++kt) {
      v16h a = frag_a_pk(&sh32[(wave * 16) * LDW + kt * 16], lane);
      v16h b = frag_b_pk(&swb[kt * 16 * LDB], nt * 16, lane);
      acc = __builtin_amdgcn_wmma_f32_16x16x32_f16(false, a, false, b,
                                                   (short)0, acc, false, false);
    }
    const unsigned ch = (unsigned)(nt * 16 + nlane);
    // batch all gathers first (8 loads in flight), then compute + scatter
    float xv[8];
#pragma unroll
    for (int r = 0; r < 8; ++r) {
      const int eloc = wave * 16 + r + 8 * half;   // same edge across 16 lanes
      xv[r] = hin[srow[eloc] + ch];                // 64B-coalesced per half-wave
    }
#pragma unroll
    for (int r = 0; r < 8; ++r) {
      const int eloc = wave * 16 + r + 8 * half;
      const float msg = fmaxf(xv[r] + acc[r], 0.f) * sew[eloc];
      atomicAdd(&agg[scol[eloc] + ch], msg);
    }
  }
}

// ---------------------------------------------------------------------------
// Layers 1,2: fused node MLP (two chained WMMA GEMMs) + BN stat accumulation.
// ---------------------------------------------------------------------------
#define TILE_N 128

__global__ __launch_bounds__(256)
void node_mlp_wmma(const float* __restrict__ x,      // N x 64
                   const float* __restrict__ agg,    // N x 64
                   const float* __restrict__ epsp,   // 1
                   const float* __restrict__ w1,     // 64 x 64
                   const float* __restrict__ b1,
                   const float* __restrict__ w2,     // 64 x 64
                   const float* __restrict__ b2,
                   float* __restrict__ outp,         // N x 64 (pre-BN)
                   float* __restrict__ stats,        // [0:64) sum, [64:128) sumsq
                   int n_nodes)
{
  __shared__ unsigned sh32[TILE_N * LDW];
  __shared__ unsigned st32[TILE_N * LDW];
  __shared__ unsigned swb[32 * LDB];
  __shared__ float sb[64];
  __shared__ float sstat[128];

  const int t = threadIdx.x;
  const int n0 = blockIdx.x * TILE_N;
  const float eps1 = 1.0f + epsp[0];

  for (int i = t; i < 128; i += 256) sstat[i] = 0.f;
  for (int i = t; i < TILE_N * 32; i += 256) {
    const int nl = i >> 5, cp = i & 31;
    const int node = n0 + nl;
    union { _Float16 h[2]; unsigned u; } pk;
#pragma unroll
    for (int s = 0; s < 2; ++s) {
      const int c = 2 * cp + s;
      float v = 0.f;
      if (node < n_nodes) v = eps1 * x[node * 64 + c] + agg[node * 64 + c];
      pk.h[s] = (_Float16)v;
    }
    sh32[nl * LDW + cp] = pk.u;
  }
  stage_w_packed(w1, swb, t);
  if (t < 64) sb[t] = b1[t];
  __syncthreads();

  const int wave = t >> 5, lane = t & 31;
  const int half = lane >> 4, nlane = lane & 15;
  _Float16* st16 = (_Float16*)st32;

  // GEMM1: u = relu(h @ W1 + b1) -> st (f16)
#pragma unroll
  for (int nt = 0; nt < 4; ++nt) {
    const float bias = sb[nt * 16 + nlane];
    v8f acc = {bias, bias, bias, bias, bias, bias, bias, bias};
#pragma unroll
    for (int kt = 0; kt < 2; ++kt) {
      v16h a = frag_a_pk(&sh32[(wave * 16) * LDW + kt * 16], lane);
      v16h b = frag_b_pk(&swb[kt * 16 * LDB], nt * 16, lane);
      acc = __builtin_amdgcn_wmma_f32_16x16x32_f16(false, a, false, b,
                                                   (short)0, acc, false, false);
    }
#pragma unroll
    for (int r = 0; r < 8; ++r) {
      const int m = wave * 16 + r + 8 * half;
      st16[m * 2 * LDW + nt * 16 + nlane] = (_Float16)fmaxf(acc[r], 0.f);
    }
  }
  __syncthreads();

  stage_w_packed(w2, swb, t);                  // overwrite with W2
  if (t < 64) sb[t] = b2[t];
  __syncthreads();

  // GEMM2: out = u @ W2 + b2, fused BN sum/sumsq
#pragma unroll
  for (int nt = 0; nt < 4; ++nt) {
    const float bias = sb[nt * 16 + nlane];
    v8f acc = {bias, bias, bias, bias, bias, bias, bias, bias};
#pragma unroll
    for (int kt = 0; kt < 2; ++kt) {
      v16h a = frag_a_pk(&st32[(wave * 16) * LDW + kt * 16], lane);
      v16h b = frag_b_pk(&swb[kt * 16 * LDB], nt * 16, lane);
      acc = __builtin_amdgcn_wmma_f32_16x16x32_f16(false, a, false, b,
                                                   (short)0, acc, false, false);
    }
    const unsigned ch = (unsigned)(nt * 16 + nlane);
    float lsum = 0.f, lsq = 0.f;
#pragma unroll
    for (int r = 0; r < 8; ++r) {
      const int node = n0 + wave * 16 + r + 8 * half;
      if (node < n_nodes) {
        const float v = acc[r];
        outp[(unsigned)node * 64u + ch] = v;
        lsum += v;
        lsq  += v * v;
      }
    }
    atomicAdd(&sstat[ch], lsum);
    atomicAdd(&sstat[64 + ch], lsq);
  }
  __syncthreads();
  if (t < 128) atomicAdd(&stats[t], sstat[t]);
}

// ---------------------------------------------------------------------------
// Layer 0 (d1=28, not WMMA-shaped; <10% of FLOPs): scalar fused edge kernel
// ---------------------------------------------------------------------------
__global__ __launch_bounds__(256)
void edge_layer0(const float* __restrict__ x, const float* __restrict__ eattr,
                 const float* __restrict__ ew,
                 const long long* __restrict__ rowi, const long long* __restrict__ coli,
                 const float* __restrict__ w1, const float* __restrict__ b1,
                 const float* __restrict__ w2, const float* __restrict__ b2,
                 float* __restrict__ agg)
{
  __shared__ float sw1[EFC * NFC], sb1[NFC], sw2[NFC * NFC], sb2[NFC];
  const int t = threadIdx.x;
  for (int i = t; i < EFC * NFC; i += 256) sw1[i] = w1[i];
  for (int i = t; i < NFC * NFC; i += 256) sw2[i] = w2[i];
  if (t < NFC) { sb1[t] = b1[t]; sb2[t] = b2[t]; }
  __syncthreads();

  const int e = blockIdx.x * 256 + t;
  if (e >= NEDGES) return;

  const float a0 = eattr[e * 3], a1 = eattr[e * 3 + 1], a2 = eattr[e * 3 + 2];
  float hid[NFC];
#pragma unroll
  for (int c = 0; c < NFC; ++c)
    hid[c] = fmaxf(sb1[c] + a0 * sw1[c] + a1 * sw1[NFC + c] + a2 * sw1[2 * NFC + c], 0.f);

  const unsigned r = (unsigned)rowi[e] * NFC, d = (unsigned)coli[e] * NFC;
  const float wgt = ew[e];
  for (int j = 0; j < NFC; ++j) {
    float ev = sb2[j];
#pragma unroll
    for (int k = 0; k < NFC; ++k) ev += hid[k] * sw2[k * NFC + j];
    const float msg = fmaxf(x[r + j] + ev, 0.f) * wgt;
    atomicAdd(&agg[d + j], msg);
  }
}

__global__ __launch_bounds__(256)
void node_layer0(const float* __restrict__ x, const float* __restrict__ agg,
                 const float* __restrict__ epsp,
                 const float* __restrict__ w1, const float* __restrict__ b1,
                 const float* __restrict__ w2, const float* __restrict__ b2,
                 float* __restrict__ outp, float* __restrict__ stats)
{
  __shared__ float sw1[NFC * NFC], sb1[NFC], sw2[NFC * DIMC], sb2[DIMC];
  __shared__ float sstat[128];
  const int t = threadIdx.x;
  for (int i = t; i < NFC * NFC;  i += 256) sw1[i] = w1[i];
  for (int i = t; i < NFC * DIMC; i += 256) sw2[i] = w2[i];
  if (t < NFC)  sb1[t] = b1[t];
  if (t < DIMC) sb2[t] = b2[t];
  for (int i = t; i < 128; i += 256) sstat[i] = 0.f;
  __syncthreads();

  const int n = blockIdx.x * 256 + t;
  const float eps1 = 1.0f + epsp[0];
  if (n < NNODES) {
    float h[NFC], u[NFC];
#pragma unroll
    for (int c = 0; c < NFC; ++c) h[c] = eps1 * x[n * NFC + c] + agg[n * NFC + c];
    for (int j = 0; j < NFC; ++j) {
      float acc = sb1[j];
#pragma unroll
      for (int k = 0; k < NFC; ++k) acc += h[k] * sw1[k * NFC + j];
      u[j] = fmaxf(acc, 0.f);
    }
    for (int j = 0; j < DIMC; ++j) {
      float acc = sb2[j];
#pragma unroll
      for (int k = 0; k < NFC; ++k) acc += u[k] * sw2[k * DIMC + j];
      outp[n * DIMC + j] = acc;
      atomicAdd(&sstat[j], acc);
      atomicAdd(&sstat[64 + j], acc * acc);
    }
  }
  __syncthreads();
  if (t < 128) atomicAdd(&stats[t], sstat[t]);
}

// ---------------------------------------------------------------------------
// BatchNorm finalize / apply, pooling, FC head
// ---------------------------------------------------------------------------
__global__ void bn_finalize(float* __restrict__ stats, float inv_n) {
  const int c = threadIdx.x;                 // 64 threads
  const float mu  = stats[c] * inv_n;
  const float var = stats[64 + c] * inv_n - mu * mu;
  stats[128 + c] = mu;
  stats[192 + c] = rsqrtf(var + 1e-5f);
}

__global__ __launch_bounds__(256)
void bn_apply(const float* __restrict__ inp, const float* __restrict__ stats,
              const float* __restrict__ gamma, const float* __restrict__ beta,
              float* __restrict__ outp, int total4) {
  const int i = blockIdx.x * 256 + threadIdx.x;      // one float4 per thread
  if (i >= total4) return;
  const int c0 = (i & 15) * 4;
  f32x4 v = *(const f32x4*)(inp + (unsigned)i * 4u);
  f32x4 o;
#pragma unroll
  for (int s = 0; s < 4; ++s) {
    const int c = c0 + s;
    o[s] = fmaxf((v[s] - stats[128 + c]) * stats[192 + c] * gamma[c] + beta[c], 0.f);
  }
  *(f32x4*)(outp + (unsigned)i * 4u) = o;
}

__global__ __launch_bounds__(256)
void pool_kernel(const float* __restrict__ h, const long long* __restrict__ batch,
                 float* __restrict__ pooled, float* __restrict__ counts) {
  // one (node, channel) pair per thread: fully coalesced, one atomic each
  const int idx = blockIdx.x * 256 + threadIdx.x;
  const int n = idx >> 6, c = idx & 63;
  if (n >= NNODES) return;
  const unsigned g = (unsigned)batch[n];
  if (c == 0) atomicAdd(&counts[g], 1.0f);
  atomicAdd(&pooled[g * DIMC + c], h[(unsigned)idx]);
}

__global__ __launch_bounds__(64)
void head_kernel(const float* __restrict__ pooled, const float* __restrict__ counts,
                 const float* __restrict__ w1, const float* __restrict__ b1,
                 const float* __restrict__ w2, const float* __restrict__ b2,
                 const float* __restrict__ w3, const float* __restrict__ b3,
                 const float* __restrict__ w4, const float* __restrict__ b4,
                 float* __restrict__ outp) {
  __shared__ float g[64], tmp[64];
  const int j = threadIdx.x, gi = blockIdx.x;
  const float cnt = fmaxf(counts[gi], 1.0f);
  g[j] = pooled[gi * 64 + j] / cnt;
  __syncthreads();

  const float* ws[3] = {w1, w2, w3};
  const float* bs[3] = {b1, b2, b3};
  for (int l = 0; l < 3; ++l) {
    float acc = bs[l][j];
    for (int k = 0; k < 64; ++k) acc += g[k] * ws[l][k * 64 + j];
    __syncthreads();
    g[j] = fmaxf(acc, 0.f);
    __syncthreads();
  }
  tmp[j] = g[j] * w4[j];
  __syncthreads();
  for (int s = 32; s > 0; s >>= 1) {
    if (j < s) tmp[j] += tmp[j + s];
    __syncthreads();
  }
  if (j == 0) outp[gi] = tmp[0] + b4[0];
}

// ---------------------------------------------------------------------------
// Launch
// ---------------------------------------------------------------------------
extern "C" void kernel_launch(void* const* d_in, const int* in_sizes, int n_in,
                              void* d_out, int out_size, void* d_ws, size_t ws_size,
                              hipStream_t stream) {
  (void)in_sizes; (void)n_in; (void)out_size; (void)ws_size;

  const float*     x     = (const float*)d_in[0];
  const long long* ei    = (const long long*)d_in[1];
  const float*     eattr = (const float*)d_in[2];
  const float*     ew    = (const float*)d_in[3];
  const long long* bat   = (const long long*)d_in[4];
  auto P = [&](int i) { return (const float*)d_in[i]; };

  const long long* rowi = ei;             // edge_index[0]
  const long long* coli = ei + NEDGES;    // edge_index[1]

  char* ws = (char*)d_ws;
  size_t off = 0;
  auto wsa = [&](size_t bytes) {
    void* p = ws + off;
    off += (bytes + 255) & ~(size_t)255;
    return p;
  };
  float* agg    = (float*)wsa((size_t)NNODES * 64 * 4);
  float* mlpout = (float*)wsa((size_t)NNODES * 64 * 4);
  float* hcur   = (float*)wsa((size_t)NNODES * 64 * 4);
  float* stats  = (float*)wsa(256 * 4);
  float* pooled = (float*)wsa((size_t)NGRAPH * 64 * 4);
  float* counts = (float*)wsa((size_t)NGRAPH * 4);

  // ---- layer 0 (d1 = 28, scalar path) ----
  hipMemsetAsync(agg, 0, (size_t)NNODES * NFC * 4, stream);
  hipMemsetAsync(stats, 0, 256 * 4, stream);
  edge_layer0<<<(NEDGES + 255) / 256, 256, 0, stream>>>(
      x, eattr, ew, rowi, coli, P(5), P(6), P(7), P(8), agg);
  node_layer0<<<(NNODES + 255) / 256, 256, 0, stream>>>(
      x, agg, P(13), P(9), P(10), P(11), P(12), mlpout, stats);
  bn_finalize<<<1, 64, 0, stream>>>(stats, 1.0f / NNODES);
  bn_apply<<<(NNODES * 16 + 255) / 256, 256, 0, stream>>>(
      mlpout, stats, P(32), P(33), hcur, NNODES * 16);

  // ---- layers 1,2 (d = 64, WMMA path) ----
  for (int l = 1; l < 3; ++l) {
    const int base = 5 + 9 * l;  // be_w1 w,b | be_w2 w,b | mlp_w1 w,b | mlp_w2 w,b | eps
    hipMemsetAsync(agg, 0, (size_t)NNODES * 64 * 4, stream);
    hipMemsetAsync(stats, 0, 256 * 4, stream);
    edge_fused_wmma<<<NEDGES / TILE_E, 256, 0, stream>>>(
        hcur, eattr, ew, rowi, coli,
        P(base + 0), P(base + 1), P(base + 2), P(base + 3), agg);
    node_mlp_wmma<<<(NNODES + TILE_N - 1) / TILE_N, 256, 0, stream>>>(
        hcur, agg, P(base + 8),
        P(base + 4), P(base + 5), P(base + 6), P(base + 7), mlpout, stats, NNODES);
    bn_finalize<<<1, 64, 0, stream>>>(stats, 1.0f / NNODES);
    bn_apply<<<(NNODES * 16 + 255) / 256, 256, 0, stream>>>(
        mlpout, stats, P(32 + 2 * l), P(33 + 2 * l), hcur, NNODES * 16);
  }

  // ---- pooling + head ----
  hipMemsetAsync(pooled, 0, (size_t)NGRAPH * 64 * 4, stream);
  hipMemsetAsync(counts, 0, (size_t)NGRAPH * 4, stream);
  pool_kernel<<<(NNODES * 64 + 255) / 256, 256, 0, stream>>>(hcur, bat, pooled, counts);
  head_kernel<<<NGRAPH, 64, 0, stream>>>(
      pooled, counts, P(38), P(39), P(40), P(41), P(42), P(43), P(44), P(45),
      (float*)d_out);
}
